// PhpNetGraph_62010737820202
// MI455X (gfx1250) — matrix operand order repaired and checked by reading, not compiled, not run on hardware
//
#include <hip/hip_runtime.h>

// ---------------------------------------------------------------------------
// Problem constants (match reference)
// ---------------------------------------------------------------------------
static constexpr int kN   = 16384;    // nodes
static constexpr int kE   = 262144;   // edges (2^18, power of two -> bitonic)
static constexpr int kNG  = 32;       // graphs
static constexpr float kNEG = -1e30f;

// ---------------------------------------------------------------------------
// WMMA / TDM types (CDNA5 wave32)
// ---------------------------------------------------------------------------
typedef __bf16 v16bf __attribute__((ext_vector_type(16)));
typedef __bf16 v8bf  __attribute__((ext_vector_type(8)));
typedef float  v8f   __attribute__((ext_vector_type(8)));
typedef unsigned int v4u __attribute__((ext_vector_type(4)));
typedef int v8i __attribute__((ext_vector_type(8)));
typedef int v4i __attribute__((ext_vector_type(4)));

__device__ __forceinline__ __bf16 f2bf(float f) { return (__bf16)f; }

// Toolchain detection for the TDM builtin arity (probe-documented difference):
//   therock-10.0 headers (amdgpu-toolchain, clang-23) ship the TDM header and
//   declare the 6-arg builtin; ROCm 7.2 (clang-22) declares the 5-arg form.
#if defined(__has_include)
#  if __has_include(<hip/amd_detail/amd_gfx1250_TDM.h>)
#    define GCN_TDM_6ARG 1
#  endif
#endif
#if __has_builtin(__builtin_amdgcn_tensor_load_to_lds)
#  define GCN_USE_TDM 1
#endif

// ---------------------------------------------------------------------------
// Small helpers
// ---------------------------------------------------------------------------
__device__ __forceinline__ void atomicMaxF(float* addr, float val) {
    int* ia = (int*)addr;
    int old = __float_as_int(*addr);
    while (__int_as_float(old) < val) {
        int assumed = old;
        old = atomicCAS(ia, assumed, __float_as_int(val));
        if (old == assumed) break;
    }
}

// float -> uint, order-preserving (ascending)
__device__ __forceinline__ unsigned floatKeyAsc(float f) {
    unsigned u = __float_as_uint(f);
    return (u & 0x80000000u) ? ~u : (u | 0x80000000u);
}

__global__ void k_fill_f(float* p, float v, int n) {
    int i = blockIdx.x * blockDim.x + threadIdx.x;
    if (i < n) p[i] = v;
}
__global__ void k_fill_i(int* p, int v, int n) {
    int i = blockIdx.x * blockDim.x + threadIdx.x;
    if (i < n) p[i] = v;
}

// ---------------------------------------------------------------------------
// Operand packing: fp32 -> zero-padded bf16 so the GEMM hot loop is guard-free
// Xbf[N][Kpad]   (row-major activations)
// Wt [Npad][Kpad] (weights transposed: column n contiguous over k)
// ---------------------------------------------------------------------------
__global__ void k_pack_x(const float* __restrict__ X, __bf16* __restrict__ Xbf,
                         int Fin, int Kpad) {
    int idx = blockIdx.x * blockDim.x + threadIdx.x;
    if (idx >= kN * Kpad) return;
    int i = idx / Kpad, k = idx - i * Kpad;
    Xbf[idx] = f2bf(k < Fin ? X[(size_t)i * Fin + k] : 0.0f);
}

__global__ void k_pack_w(const float* __restrict__ W, __bf16* __restrict__ Wt,
                         int Fin, int Fout, int Kpad, int Npad) {
    int idx = blockIdx.x * blockDim.x + threadIdx.x;
    if (idx >= Npad * Kpad) return;
    int n = idx / Kpad, k = idx - n * Kpad;
    Wt[idx] = f2bf((k < Fin && n < Fout) ? W[(size_t)k * Fout + n] : 0.0f);
}

// ---------------------------------------------------------------------------
// GEMM: H[N,Fout] = Xbf[N,Kpad] @ Wt^T, bf16 WMMA, fp32 accumulate.
// Block = 256 threads = 8 wave32 waves; block tile 16 rows x 128 cols.
// A tile (16x32 bf16) lands in LDS via the Tensor Data Mover (wave 0 issues a
// 2D D# tile descriptor, waits on TENSORcnt); fragments are aligned b128 ops.
// ---------------------------------------------------------------------------
__global__ void __launch_bounds__(256)
k_gemm_wmma(const __bf16* __restrict__ Xbf, const __bf16* __restrict__ Wt,
            float* __restrict__ H, int Kpad, int Fout) {
    __shared__ __align__(16) __bf16 Atile[16 * 32];

    const int tid  = threadIdx.x;
    const int wave = tid >> 5;
    const int lane = tid & 31;
    const int lm   = lane & 15;     // row-in-tile (A) / col-in-tile (B,C)
    const int lh   = lane >> 4;     // lane-half select
    const int row0 = blockIdx.x * 16;
    const int col0 = blockIdx.y * 128 + wave * 16;
    const int nn   = col0 + lm;
    const __bf16* wtcol = &Wt[(size_t)nn * Kpad];

    v8f acc = {};
    for (int k0 = 0; k0 < Kpad; k0 += 32) {
#if defined(GCN_USE_TDM)
        if (wave == 0) {
            // 2D tile: tile_dim0=32 (contiguous k, bf16), tile_dim1=16 rows,
            // row stride = Kpad elements; global_addr = tile start.
            const __bf16* tile = &Xbf[(size_t)row0 * Kpad + k0];
            unsigned long long ga = (unsigned long long)(uintptr_t)tile;
            unsigned ldsOff = (unsigned)(unsigned long long)(uintptr_t)Atile;
            v4u g0 = { 1u,                                    // count=1
                       ldsOff,                                // lds_addr
                       (unsigned)(ga & 0xffffffffu),          // global_addr lo
                       ((unsigned)(ga >> 32) & 0x01ffffffu)   // global_addr hi
                           | (2u << 30) };                    // type=2 (image)
            v8i g1 = { (int)(1u << 16),        // data_size=1 (2 bytes)
                       (int)(32u << 16),       // tensor_dim0=32 (lo16<<16)
                       (int)(16u << 16),       // tensor_dim1=16 (lo16<<16)
                       (int)(32u << 16),       // tile_dim0=32
                       16,                     // tile_dim1=16, tile_dim2=0
                       (int)Kpad,              // tensor_dim0_stride lo32
                       0, 0 };                 // stride0 hi / stride1
            v4i gz = {0, 0, 0, 0};
#if defined(GCN_TDM_6ARG)
            v8i gz8 = {0, 0, 0, 0, 0, 0, 0, 0};
            __builtin_amdgcn_tensor_load_to_lds(g0, g1, gz, gz, gz8, 0);
#else
            __builtin_amdgcn_tensor_load_to_lds(g0, g1, gz, gz, 0);
#endif
            __builtin_amdgcn_s_wait_tensorcnt(0);
        }
#else
        {   // vectorized cooperative fallback: 256 threads x one b32
            int r = tid >> 4, kk = (tid & 15) << 1;
            *(unsigned*)&Atile[r * 32 + kk] =
                *(const unsigned*)&Xbf[(size_t)(row0 + r) * Kpad + k0 + kk];
        }
#endif
        __syncthreads();
        if (k0 + 32 < Kpad)   // gfx1250 global_prefetch_b8 for next B tile
            __builtin_prefetch(&wtcol[k0 + 32 + (lh << 4)], 0, 1);

        // A fragment: lane holds row lm; K = 16*lh + e
        //  -> two aligned 16B LDS chunks at k = lh*16 and k = lh*16 + 8
        const v8bf* arow = (const v8bf*)&Atile[lm * 32];
        v8bf alo = arow[2 * lh];
        v8bf ahi = arow[2 * lh + 1];
        v16bf a = __builtin_shufflevector(alo, ahi, 0, 1, 2, 3, 4, 5, 6, 7,
                                          8, 9, 10, 11, 12, 13, 14, 15);
        // B fragment: lane holds column nn; K = 16*lh + e -> one 32B load
        v16bf b = *(const v16bf*)&wtcol[k0 + (lh << 4)];

        acc = __builtin_amdgcn_wmma_f32_16x16x32_bf16(
            /*neg_a=*/false, a, /*neg_b=*/false, b,
            /*c_mod=*/(short)0, acc, /*reuse_a=*/false, /*reuse_b=*/false);
        __syncthreads();
    }
    if (nn < Fout) {
#pragma unroll
        for (int r = 0; r < 8; ++r)    // C layout: VGPR r -> M = 8*lh + r
            H[(size_t)(row0 + (lh << 3) + r) * Fout + nn] = acc[r];
    }
}

// ---------------------------------------------------------------------------
// GCN normalization pieces
// ---------------------------------------------------------------------------
__global__ void k_init_edges(const int* __restrict__ ei, int* src, int* dst, int* em) {
    int e = blockIdx.x * blockDim.x + threadIdx.x;
    if (e < kE) { src[e] = ei[e]; dst[e] = ei[kE + e]; em[e] = 1; }
}

__global__ void k_deg(const int* src, const int* dst, const int* em,
                      float* deg, int* hasself) {
    int e = blockIdx.x * blockDim.x + threadIdx.x;
    if (e >= kE || !em[e]) return;
    atomicAdd(&deg[dst[e]], 1.0f);
    if (src[e] == dst[e]) hasself[dst[e]] = 1;   // race-safe (same value)
}

__global__ void k_dinv(const float* deg, const int* hasself,
                       float* dinv, float* coef) {
    int i = blockIdx.x * blockDim.x + threadIdx.x;
    if (i >= kN) return;
    float hs = hasself[i] ? 1.0f : 0.0f;
    float dg = deg[i] + (1.0f - hs);
    dinv[i] = rsqrtf(dg);
    coef[i] = (1.0f - hs) / dg;
}

__global__ void k_norm(const int* src, const int* dst, const int* em,
                       const float* dinv, float* nrm) {
    int e = blockIdx.x * blockDim.x + threadIdx.x;
    if (e >= kE) return;
    nrm[e] = em[e] ? dinv[src[e]] * dinv[dst[e]] : 0.0f;
}

// agg[dst] += norm * h[src] ; one thread per (edge, feature)
__global__ void k_agg(const float* __restrict__ h, const int* src, const int* dst,
                      const int* em, const float* nrm, float* agg, int F) {
    int idx = blockIdx.x * blockDim.x + threadIdx.x;
    if (idx >= kE * F) return;
    int e = idx / F, f = idx - e * F;
    if (!em[e]) return;
    atomicAdd(&agg[(size_t)dst[e] * F + f], nrm[e] * h[(size_t)src[e] * F + f]);
}

// out = agg + coef[i]*h + b
__global__ void k_gcn_finish(float* agg, const float* h, const float* coef,
                             const float* b, int F) {
    int idx = blockIdx.x * blockDim.x + threadIdx.x;
    if (idx >= kN * F) return;
    int i = idx / F, f = idx - i * F;
    agg[idx] += coef[i] * h[idx] + b[f];
}

// ---------------------------------------------------------------------------
// EdgePooling
// ---------------------------------------------------------------------------
// raw[e] = dot(concat(h[src], h[dst]), Pw) + Pb ; one wave32 per edge
__global__ void __launch_bounds__(256)
k_edge_score(const float* __restrict__ h, const int* src, const int* dst,
             const float* __restrict__ Pw, const float* Pb, float* raw, int F) {
    int e = blockIdx.x * 8 + (threadIdx.x >> 5);
    int lane = threadIdx.x & 31;
    if (e >= kE) return;
    int s = src[e], d = dst[e];
    float acc = 0.0f;
    for (int f = lane; f < F; f += 32)
        acc += h[(size_t)s * F + f] * Pw[f] + h[(size_t)d * F + f] * Pw[F + f];
    for (int off = 16; off; off >>= 1) acc += __shfl_xor(acc, off, 32);
    if (lane == 0) raw[e] = acc + Pb[0];
}

__global__ void k_smax(const int* dst, const int* em, const float* raw, float* mx) {
    int e = blockIdx.x * blockDim.x + threadIdx.x;
    if (e >= kE || !em[e]) return;
    atomicMaxF(&mx[dst[e]], raw[e]);
}

__global__ void k_sexp(const int* dst, const int* em, const float* raw,
                       const float* mx, float* exb, float* den) {
    int e = blockIdx.x * blockDim.x + threadIdx.x;
    if (e >= kE) return;
    if (em[e]) {
        float ex = expf(raw[e] - mx[dst[e]]);
        exb[e] = ex;
        atomicAdd(&den[dst[e]], ex);
    } else {
        exb[e] = 0.0f;
    }
}

// score = softmax + 0.1; pack stable descending argsort key
__global__ void k_score_pack(const int* dst, const int* em, const float* exb,
                             const float* den, float* score,
                             unsigned long long* skey) {
    int e = blockIdx.x * blockDim.x + threadIdx.x;
    if (e >= kE) return;
    float sc = (em[e] ? exb[e] / fmaxf(den[dst[e]], 1e-16f) : 0.0f) + 0.1f;
    score[e] = sc;
    float keyf = em[e] ? sc : kNEG;
    unsigned inv = ~floatKeyAsc(keyf);          // ascending sort => desc. score
    skey[e] = ((unsigned long long)inv << 32) | (unsigned)e;
}

// classic global bitonic stage; E is a power of two
__global__ void k_bitonic(unsigned long long* a, int j, int k) {
    int i = blockIdx.x * blockDim.x + threadIdx.x;
    int ixj = i ^ j;
    if (ixj > i) {
        unsigned long long x = a[i], y = a[ixj];
        bool asc = ((i & k) == 0);
        if ((x > y) == asc) { a[i] = y; a[ixj] = x; }
    }
}

__global__ void k_node_init(const int* nvalid, int* cluster, float* nscore,
                            int* marked) {
    int i = blockIdx.x * blockDim.x + threadIdx.x;
    if (i >= kN) return;
    cluster[i] = i;
    nscore[i]  = 1.0f;
    marked[i]  = nvalid[i] ? 0 : 1;
}

// sequential greedy matching in descending-score order (matches lax.scan)
__global__ void k_match(const unsigned long long* skey, const int* src,
                        const int* dst, const int* em, int* marked, int* chosen) {
    if (threadIdx.x | blockIdx.x) return;
    for (int i = 0; i < kE; ++i) {
        int e = (int)(skey[i] & 0xffffffffu);
        int s = src[e], d = dst[e];
        int ok = em[e] && !marked[s] && !marked[d];
        if (ok) { marked[s] = 1; marked[d] = 1; }
        chosen[e] = ok;
    }
}

__global__ void k_contract(const int* src, const int* dst, const int* chosen,
                           const float* score, int* cluster, float* nscore,
                           int* nvalid) {
    int e = blockIdx.x * blockDim.x + threadIdx.x;
    if (e >= kE || !chosen[e]) return;
    int s = src[e], d = dst[e];
    int rep = min(s, d), oth = max(s, d);
    cluster[oth] = rep;              // matching => no write races
    nscore[rep]  = score[e];
    if (s != d) nvalid[oth] = 0;     // absorbed
}

__global__ void k_pool_scatter(const float* __restrict__ h, const int* cluster,
                               float* out, int F) {
    int idx = blockIdx.x * blockDim.x + threadIdx.x;
    if (idx >= kN * F) return;
    int i = idx / F, f = idx - i * F;
    atomicAdd(&out[(size_t)cluster[i] * F + f], h[idx]);
}

__global__ void k_pool_finish(float* x, const float* nscore, const int* nvalid, int F) {
    int idx = blockIdx.x * blockDim.x + threadIdx.x;
    if (idx >= kN * F) return;
    int i = idx / F;
    float v = fmaxf(x[idx] * nscore[i], 0.0f);   // * node_score, relu
    x[idx] = nvalid[i] ? v : 0.0f;               // * new_valid
}

// pack edge-coalesce key: (ns*N + nd) << 32 | e  (stable ascending)
__global__ void k_remap_key(const int* src, const int* dst, const int* em,
                            const int* cluster, unsigned long long* skey) {
    int e = blockIdx.x * blockDim.x + threadIdx.x;
    if (e >= kE) return;
    unsigned long long hi;
    if (em[e]) {
        int ns = cluster[src[e]], nd = cluster[dst[e]];
        hi = (unsigned long long)((unsigned)ns * (unsigned)kN + (unsigned)nd);
    } else {
        hi = (unsigned long long)kN * (unsigned long long)kN;  // 2^28 sentinel
    }
    skey[e] = (hi << 32) | (unsigned)e;
}

__global__ void k_edges_rebuild(const unsigned long long* skey, int* src,
                                int* dst, int* em) {
    int i = blockIdx.x * blockDim.x + threadIdx.x;
    if (i >= kE) return;
    unsigned hi = (unsigned)(skey[i] >> 32);
    bool valid = hi < (unsigned)(kN) * (unsigned)(kN);
    bool dup = (i > 0) && ((unsigned)(skey[i - 1] >> 32) == hi);
    em[i]  = (valid && !dup) ? 1 : 0;
    src[i] = valid ? (int)(hi >> 14) : 0;       // kN == 2^14
    dst[i] = valid ? (int)(hi & 16383) : 0;
}

// ---------------------------------------------------------------------------
// Readout
// ---------------------------------------------------------------------------
__global__ void k_gmax(const float* __restrict__ h, const int* batch,
                       const int* nvalid, float* g) {
    int idx = blockIdx.x * blockDim.x + threadIdx.x;
    if (idx >= kN * 600) return;
    int i = idx / 600, f = idx - i * 600;
    if (!nvalid[i]) return;
    atomicMaxF(&g[(size_t)batch[i] * 600 + f], h[idx]);
}

__global__ void k_mlp(const float* __restrict__ in, const float* __restrict__ Wm,
                      const float* __restrict__ bm, float* out, int R, int Ci, int Co) {
    int idx = blockIdx.x * blockDim.x + threadIdx.x;
    if (idx >= R * Co) return;
    int r = idx / Co, c = idx - r * Co;
    float s = bm[c];
    for (int k = 0; k < Ci; ++k) s += in[(size_t)r * Ci + k] * Wm[(size_t)k * Co + c];
    out[idx] = fmaxf(s, 0.0f);
}

__global__ void k_out(const float* z2, const float* W3, const float* b3, float* out) {
    int r = threadIdx.x;
    if (r >= kNG) return;
    float v[4], m = -1e30f;
    for (int c = 0; c < 4; ++c) {
        float s = b3[c];
        for (int k = 0; k < 20; ++k) s += z2[r * 20 + k] * W3[k * 4 + c];
        s = fmaxf(s, 0.0f);
        v[c] = s; m = fmaxf(m, s);
    }
    float den = 0.0f;
    for (int c = 0; c < 4; ++c) { v[c] = expf(v[c] - m); den += v[c]; }
    for (int c = 0; c < 4; ++c) out[r * 4 + c] = v[c] / den;
}

// ---------------------------------------------------------------------------
// Orchestration
// ---------------------------------------------------------------------------
extern "C" void kernel_launch(void* const* d_in, const int* in_sizes, int n_in,
                              void* d_out, int out_size, void* d_ws, size_t ws_size,
                              hipStream_t stream) {
    (void)in_sizes; (void)n_in; (void)out_size; (void)ws_size;

    const float* x     = (const float*)d_in[0];
    const int*   ei    = (const int*)d_in[1];
    const int*   batch = (const int*)d_in[2];
    const float* W[3]  = {(const float*)d_in[3], (const float*)d_in[5], (const float*)d_in[7]};
    const float* B[3]  = {(const float*)d_in[4], (const float*)d_in[6], (const float*)d_in[8]};
    const float* Pw[3] = {(const float*)d_in[9], (const float*)d_in[11], (const float*)d_in[13]};
    const float* Pb[3] = {(const float*)d_in[10], (const float*)d_in[12], (const float*)d_in[14]};
    const float* L1w = (const float*)d_in[15]; const float* L1b = (const float*)d_in[16];
    const float* L2w = (const float*)d_in[17]; const float* L2b = (const float*)d_in[18];
    const float* L3w = (const float*)d_in[19]; const float* L3b = (const float*)d_in[20];

    // deterministic workspace carve
    char* base = (char*)d_ws;
    size_t off = 0;
    auto carve = [&](size_t bytes) -> void* {
        void* p = base + off;
        off = (off + bytes + 255) & ~(size_t)255;
        return p;
    };
    unsigned long long* skey = (unsigned long long*)carve((size_t)kE * 8);
    float* bufA = (float*)carve((size_t)kN * 600 * 4);   // pooled features
    float* bufB = (float*)carve((size_t)kN * 600 * 4);   // h = xW
    float* bufC = (float*)carve((size_t)kN * 600 * 4);   // agg / gcn output
    __bf16* Xbf = (__bf16*)carve((size_t)kN * 448 * 2);  // packed activations
    __bf16* Wt  = (__bf16*)carve((size_t)640 * 448 * 2); // packed weights
    int*   src  = (int*)carve((size_t)kE * 4);
    int*   dst  = (int*)carve((size_t)kE * 4);
    int*   em   = (int*)carve((size_t)kE * 4);
    float* raw  = (float*)carve((size_t)kE * 4);
    float* exb  = (float*)carve((size_t)kE * 4);
    float* scr  = (float*)carve((size_t)kE * 4);
    int*   chosen = (int*)carve((size_t)kE * 4);
    float* nrm  = (float*)carve((size_t)kE * 4);
    float* mx   = (float*)carve((size_t)kN * 4);
    float* den  = (float*)carve((size_t)kN * 4);
    float* deg  = (float*)carve((size_t)kN * 4);
    int*   hasself = (int*)carve((size_t)kN * 4);
    float* dinv = (float*)carve((size_t)kN * 4);
    float* coef = (float*)carve((size_t)kN * 4);
    int*   marked  = (int*)carve((size_t)kN * 4);
    int*   cluster = (int*)carve((size_t)kN * 4);
    float* nscore  = (float*)carve((size_t)kN * 4);
    int*   nvalid  = (int*)carve((size_t)kN * 4);
    float* g  = (float*)carve((size_t)kNG * 600 * 4);
    float* z1 = (float*)carve((size_t)kNG * 200 * 4);
    float* z2 = (float*)carve((size_t)kNG * 20 * 4);

    const int EB = kE / 256;           // edge-grid
    const int NB = (kN + 255) / 256;   // node-grid
    auto bitonic = [&](unsigned long long* a) {
        for (int k = 2; k <= kE; k <<= 1)
            for (int j = k >> 1; j > 0; j >>= 1)
                k_bitonic<<<kE / 256, 256, 0, stream>>>(a, j, k);
    };

    k_init_edges<<<EB, 256, 0, stream>>>(ei, src, dst, em);
    k_fill_i<<<NB, 256, 0, stream>>>(nvalid, 1, kN);

    const int FinL[3]  = {107, 200, 400};
    const int FoutL[3] = {200, 400, 600};
    const float* Xcur = x;

    for (int L = 0; L < 3; ++L) {
        const int Fi = FinL[L], Fo = FoutL[L];
        const int Kpad = (Fi + 31) & ~31;          // 128, 224, 416
        const int Npad = ((Fo + 127) / 128) * 128; // 256, 512, 640
        const int NF = kN * Fo, EF = kE * Fo;

        // ---- GCNConv ----
        k_pack_x<<<(kN * Kpad + 255) / 256, 256, 0, stream>>>(Xcur, Xbf, Fi, Kpad);
        k_pack_w<<<(Npad * Kpad + 255) / 256, 256, 0, stream>>>(W[L], Wt, Fi, Fo, Kpad, Npad);
        dim3 gg(kN / 16, Npad / 128);
        k_gemm_wmma<<<gg, 256, 0, stream>>>(Xbf, Wt, bufB, Kpad, Fo);
        k_fill_f<<<NB, 256, 0, stream>>>(deg, 0.0f, kN);
        k_fill_i<<<NB, 256, 0, stream>>>(hasself, 0, kN);
        k_fill_f<<<(NF + 255) / 256, 256, 0, stream>>>(bufC, 0.0f, NF);
        k_deg<<<EB, 256, 0, stream>>>(src, dst, em, deg, hasself);
        k_dinv<<<NB, 256, 0, stream>>>(deg, hasself, dinv, coef);
        k_norm<<<EB, 256, 0, stream>>>(src, dst, em, dinv, nrm);
        k_agg<<<(EF + 255) / 256, 256, 0, stream>>>(bufB, src, dst, em, nrm, bufC, Fo);
        k_gcn_finish<<<(NF + 255) / 256, 256, 0, stream>>>(bufC, bufB, coef, B[L], Fo);

        // ---- EdgePooling ----
        k_edge_score<<<kE / 8, 256, 0, stream>>>(bufC, src, dst, Pw[L], Pb[L], raw, Fo);
        k_fill_f<<<NB, 256, 0, stream>>>(mx, kNEG, kN);
        k_fill_f<<<NB, 256, 0, stream>>>(den, 0.0f, kN);
        k_smax<<<EB, 256, 0, stream>>>(dst, em, raw, mx);
        k_sexp<<<EB, 256, 0, stream>>>(dst, em, raw, mx, exb, den);
        k_score_pack<<<EB, 256, 0, stream>>>(dst, em, exb, den, scr, skey);
        bitonic(skey);                                   // argsort(-score), stable
        k_node_init<<<NB, 256, 0, stream>>>(nvalid, cluster, nscore, marked);
        k_match<<<1, 1, 0, stream>>>(skey, src, dst, em, marked, chosen);
        k_contract<<<EB, 256, 0, stream>>>(src, dst, chosen, scr, cluster, nscore, nvalid);
        k_fill_f<<<(NF + 255) / 256, 256, 0, stream>>>(bufA, 0.0f, NF);
        k_pool_scatter<<<(NF + 255) / 256, 256, 0, stream>>>(bufC, cluster, bufA, Fo);
        k_pool_finish<<<(NF + 255) / 256, 256, 0, stream>>>(bufA, nscore, nvalid, Fo);
        k_remap_key<<<EB, 256, 0, stream>>>(src, dst, em, cluster, skey);
        bitonic(skey);                                   // edge coalesce sort
        k_edges_rebuild<<<EB, 256, 0, stream>>>(skey, src, dst, em);

        Xcur = bufA;
    }

    // ---- readout: masked segment-max over graphs + MLP + softmax ----
    k_fill_f<<<(kNG * 600 + 255) / 256, 256, 0, stream>>>(g, kNEG, kNG * 600);
    k_gmax<<<(kN * 600 + 255) / 256, 256, 0, stream>>>(bufA, batch, nvalid, g);
    k_mlp<<<(kNG * 200 + 255) / 256, 256, 0, stream>>>(g, L1w, L1b, z1, kNG, 600, 200);
    k_mlp<<<(kNG * 20 + 255) / 256, 256, 0, stream>>>(z1, L2w, L2b, z2, kNG, 200, 20);
    k_out<<<1, 32, 0, stream>>>(z2, L3w, L3b, (float*)d_out);
}
